// MultiHeadAttention_41145786695774
// MI455X (gfx1250) — matrix operand order repaired and checked by reading, not compiled, not run on hardware
//
#include <hip/hip_runtime.h>
#include <hip/hip_bf16.h>

#define EMBED 1024
#define NHEADS 16
#define HDIM 64
#define BATCH 4
#define SEQ 2048

typedef __attribute__((ext_vector_type(16))) __bf16 bf16x16;
typedef __attribute__((ext_vector_type(8)))  __bf16 bf16x8;
typedef __attribute__((ext_vector_type(4)))  __bf16 bf16x4;
typedef __attribute__((ext_vector_type(8))) float  f32x8;
typedef __attribute__((ext_vector_type(4))) float  f32x4;

// ---- CDNA5 async global->LDS copy (ASYNCcnt path) --------------------------
// LDS aperture maps flat addr[31:0] directly to the LDS offset (ISA 10.2), so
// truncating the generic pointer gives the dsaddr VGPR for the async op.
__device__ __forceinline__ void async_copy_b128(void* lds_dst, const void* gsrc) {
    unsigned loff = (unsigned)(uintptr_t)lds_dst;
    asm volatile("global_load_async_to_lds_b128 %0, %1, off"
                 :: "v"(loff), "v"(gsrc) : "memory");
}
__device__ __forceinline__ void wait_async() {
    asm volatile("s_wait_asynccnt 0x0" ::: "memory");
}

// ---- WMMA helpers (CDNA5 layouts, wave32) ----------------------------------
// A fragment: 16x32 bf16 row-major; lane L: m=L&15, hf=L>>4; K = hf*8+0..7, 16+hf*8+0..7.
__device__ __forceinline__ bf16x16 load_frag_a(const __bf16* base, int ld, int lane) {
    int m = lane & 15, hf = lane >> 4;
    const __bf16* r = base + (size_t)m * ld;
    bf16x8 lo = *(const bf16x8*)(r + hf * 8);
    bf16x8 hi = *(const bf16x8*)(r + 16 + hf * 8);
    return __builtin_shufflevector(lo, hi, 0,1,2,3,4,5,6,7,8,9,10,11,12,13,14,15);
}
// B fragment from TRANSPOSED tile Bt[n][k]; lane L: n=L&15, hf=L>>4; K = hf*8+0..7, 16+hf*8+0..7.
__device__ __forceinline__ bf16x16 load_frag_b(const __bf16* baseT, int ld, int lane) {
    int n = lane & 15, hf = lane >> 4;
    const __bf16* r = baseT + (size_t)n * ld;
    bf16x8 lo = *(const bf16x8*)(r + hf * 8);
    bf16x8 hi = *(const bf16x8*)(r + 16 + hf * 8);
    return __builtin_shufflevector(lo, hi, 0,1,2,3,4,5,6,7,8,9,10,11,12,13,14,15);
}
__device__ __forceinline__ f32x8 wmma_bf16(bf16x16 a, bf16x16 b, f32x8 c) {
    return __builtin_amdgcn_wmma_f32_16x16x32_bf16(false, a, false, b, (short)0, c, false, false);
}

// ---- Kernel 1: fused QKV projection ---------------------------------------
// C[8192,3072] = x[8192,1024] @ w_qkv[1024,3072] + b_qkv
// Block tile 128x128, K-stage 64; wave owns 32x64 (8 accumulators, 16 WMMA/stage).
// Output scattered into bf16 Q/K/V workspace [B,H,S,D]; Q pre-scaled by 1/sqrt(D).
__global__ __launch_bounds__(256)
void qkv_proj_kernel(const float* __restrict__ x, const float* __restrict__ w,
                     const float* __restrict__ bias, __bf16* __restrict__ qkv_ws)
{
    __shared__ __bf16 As[128][72];    // A tile 128x64 (padded)
    __shared__ __bf16 Bts[128][72];   // B tile transposed [n][k], 128x64
    const int tid = threadIdx.x, lane = tid & 31, wave = tid >> 5;
    const int wm = wave & 3, wn = wave >> 2;        // 4x2 wave grid
    const int m0 = blockIdx.y * 128, n0 = blockIdx.x * 128;
    f32x8 acc[2][4] = {};

    for (int k0 = 0; k0 < EMBED; k0 += 64) {
        // stage A: 128x64 fp32 -> bf16 (8 float4 per thread)
        #pragma unroll
        for (int i = 0; i < 8; ++i) {
            int idx4 = tid + i * 256;                    // 2048 float4 chunks
            int row = idx4 >> 4, c4 = (idx4 & 15) << 2;
            f32x4 v = *(const f32x4*)(x + (size_t)(m0 + row) * EMBED + k0 + c4);
            bf16x4 bv = { (__bf16)v[0], (__bf16)v[1], (__bf16)v[2], (__bf16)v[3] };
            *(bf16x4*)&As[row][c4] = bv;
        }
        // stage B transposed: 64x128 fp32 -> Bts[n][k]
        #pragma unroll
        for (int i = 0; i < 8; ++i) {
            int idx4 = tid + i * 256;                    // 2048 float4 chunks
            int kr = idx4 >> 5, c4 = (idx4 & 31) << 2;
            f32x4 v = *(const f32x4*)(w + (size_t)(k0 + kr) * 3072 + n0 + c4);
            #pragma unroll
            for (int e = 0; e < 4; ++e) Bts[c4 + e][kr] = (__bf16)v[e];
        }
        if (k0 + 64 < EMBED) {                           // global_prefetch_b8 next stage
            __builtin_prefetch(x + (size_t)(m0 + (tid >> 1)) * EMBED + k0 + 64 + (tid & 1) * 32, 0, 3);
            __builtin_prefetch(w + (size_t)(k0 + 64 + (tid >> 2)) * 3072 + n0 + (tid & 3) * 32, 0, 3);
        }
        __syncthreads();
        #pragma unroll
        for (int kk = 0; kk < 64; kk += 32) {
            // materialize all fragments first -> bulk ds_load issue, staged waits
            bf16x16 a0 = load_frag_a(&As[wm * 32][kk], 72, lane);
            bf16x16 a1 = load_frag_a(&As[wm * 32 + 16][kk], 72, lane);
            bf16x16 b0 = load_frag_b(&Bts[wn * 64 + 0][kk], 72, lane);
            bf16x16 b1 = load_frag_b(&Bts[wn * 64 + 16][kk], 72, lane);
            bf16x16 b2 = load_frag_b(&Bts[wn * 64 + 32][kk], 72, lane);
            bf16x16 b3 = load_frag_b(&Bts[wn * 64 + 48][kk], 72, lane);
            acc[0][0] = wmma_bf16(a0, b0, acc[0][0]);
            acc[1][0] = wmma_bf16(a1, b0, acc[1][0]);
            acc[0][1] = wmma_bf16(a0, b1, acc[0][1]);
            acc[1][1] = wmma_bf16(a1, b1, acc[1][1]);
            acc[0][2] = wmma_bf16(a0, b2, acc[0][2]);
            acc[1][2] = wmma_bf16(a1, b2, acc[1][2]);
            acc[0][3] = wmma_bf16(a0, b3, acc[0][3]);
            acc[1][3] = wmma_bf16(a1, b3, acc[1][3]);
        }
        __syncthreads();
    }

    // epilogue: bias, scatter to [which][B,H,S,D]
    const int hf = lane >> 4, nn = lane & 15;
    const size_t seg = (size_t)BATCH * NHEADS * SEQ * HDIM;
    #pragma unroll
    for (int mt = 0; mt < 2; ++mt) {
        #pragma unroll
        for (int nt = 0; nt < 4; ++nt) {
            #pragma unroll
            for (int r = 0; r < 8; ++r) {
                int m = m0 + wm * 32 + mt * 16 + r + 8 * hf;
                int n = n0 + wn * 64 + nt * 16 + nn;
                float v = acc[mt][nt][r] + bias[n];
                int which = n >> 10;                     // 0=Q 1=K 2=V
                int rn = n & 1023;
                int h = rn >> 6, d = rn & 63;
                if (which == 0) v *= 0.125f;             // 1/sqrt(64) folded into Q
                int bb = m >> 11, s = m & 2047;
                qkv_ws[seg * which + (((size_t)(bb * NHEADS + h)) * SEQ + s) * HDIM + d] = (__bf16)v;
            }
        }
    }
}

// ---- Kernel 2: flash attention --------------------------------------------
__global__ __launch_bounds__(256)
void attn_kernel(const __bf16* __restrict__ qkv_ws, __bf16* __restrict__ attn_ws)
{
    __shared__ __bf16 Qs[128][72];       // Q tile [q][d]
    __shared__ __bf16 Ks[64][72];        // K tile [key][d]
    __shared__ __bf16 Vts[64][72];       // V tile transposed [d][key]
    __shared__ __bf16 Ps[8][16][72];     // per-wave P tile [q][key]
    const int tid = threadIdx.x, lane = tid & 31, wave = tid >> 5;
    const int q0 = blockIdx.x * 128;
    const int h = blockIdx.y, b = blockIdx.z;
    const size_t seg = (size_t)BATCH * NHEADS * SEQ * HDIM;
    const size_t bh = ((size_t)b * NHEADS + h) * SEQ;
    const __bf16* Qg = qkv_ws + (bh + q0) * HDIM;
    const __bf16* Kg = qkv_ws + seg + bh * HDIM;
    const __bf16* Vg = qkv_ws + 2 * seg + bh * HDIM;

    // stage Q once via async copies (pure copy, bf16)
    #pragma unroll
    for (int i = 0; i < 4; ++i) {
        int idx8 = tid + i * 256;
        int row = idx8 >> 3, c8 = (idx8 & 7) << 3;
        async_copy_b128(&Qs[row][c8], Qg + (size_t)row * HDIM + c8);
    }
    wait_async();

    float mrow[8], lrow[8];
    #pragma unroll
    for (int r = 0; r < 8; ++r) { mrow[r] = -1e30f; lrow[r] = 0.f; }
    f32x8 oacc[4] = {};
    const int hf = lane >> 4, nn = lane & 15;

    for (int j0 = 0; j0 < SEQ; j0 += 64) {
        __syncthreads();                  // previous iter's readers done (also covers Q)
        #pragma unroll
        for (int i = 0; i < 2; ++i) {
            int idx8 = tid + i * 256;
            int row = idx8 >> 3, c8 = (idx8 & 7) << 3;
            // K: pure copy -> async to LDS
            async_copy_b128(&Ks[row][c8], Kg + (size_t)(j0 + row) * HDIM + c8);
            // V: transpose while staging (VALU path)
            bf16x8 vv = *(const bf16x8*)(Vg + (size_t)(j0 + row) * HDIM + c8);
            #pragma unroll
            for (int e = 0; e < 8; ++e) Vts[c8 + e][row] = vv[e];
        }
        wait_async();
        __syncthreads();

        // S = Q K^T (Q pre-scaled): 16 q-rows x 64 keys per wave
        f32x8 sacc[4] = {};
        #pragma unroll
        for (int kk = 0; kk < 2; ++kk) {
            bf16x16 a  = load_frag_a(&Qs[wave * 16][kk * 32], 72, lane);
            bf16x16 k0f = load_frag_b(&Ks[0][kk * 32], 72, lane);
            bf16x16 k1f = load_frag_b(&Ks[16][kk * 32], 72, lane);
            bf16x16 k2f = load_frag_b(&Ks[32][kk * 32], 72, lane);
            bf16x16 k3f = load_frag_b(&Ks[48][kk * 32], 72, lane);
            sacc[0] = wmma_bf16(a, k0f, sacc[0]);
            sacc[1] = wmma_bf16(a, k1f, sacc[1]);
            sacc[2] = wmma_bf16(a, k2f, sacc[2]);
            sacc[3] = wmma_bf16(a, k3f, sacc[3]);
        }

        // streaming softmax; lane group {hf} holds row r + 8*hf, col = lane&15
        #pragma unroll
        for (int r = 0; r < 8; ++r) {
            float tmax = fmaxf(fmaxf(sacc[0][r], sacc[1][r]), fmaxf(sacc[2][r], sacc[3][r]));
            #pragma unroll
            for (int off = 1; off <= 8; off <<= 1)
                tmax = fmaxf(tmax, __shfl_xor(tmax, off, 32));
            float mnew = fmaxf(mrow[r], tmax);
            float alpha = __expf(mrow[r] - mnew);
            mrow[r] = mnew;
            float rs = 0.f;
            #pragma unroll
            for (int kt = 0; kt < 4; ++kt) {
                float p = __expf(sacc[kt][r] - mnew);
                sacc[kt][r] = p;
                rs += p;
            }
            #pragma unroll
            for (int off = 1; off <= 8; off <<= 1)
                rs += __shfl_xor(rs, off, 32);
            lrow[r] = lrow[r] * alpha + rs;
            #pragma unroll
            for (int nt = 0; nt < 4; ++nt) oacc[nt][r] *= alpha;
            #pragma unroll
            for (int kt = 0; kt < 4; ++kt)
                Ps[wave][r + 8 * hf][kt * 16 + nn] = (__bf16)sacc[kt][r];
        }

        // O += P @ V
        #pragma unroll
        for (int kk = 0; kk < 2; ++kk) {
            bf16x16 a  = load_frag_a(&Ps[wave][0][kk * 32], 72, lane);
            bf16x16 v0f = load_frag_b(&Vts[0][kk * 32], 72, lane);
            bf16x16 v1f = load_frag_b(&Vts[16][kk * 32], 72, lane);
            bf16x16 v2f = load_frag_b(&Vts[32][kk * 32], 72, lane);
            bf16x16 v3f = load_frag_b(&Vts[48][kk * 32], 72, lane);
            oacc[0] = wmma_bf16(a, v0f, oacc[0]);
            oacc[1] = wmma_bf16(a, v1f, oacc[1]);
            oacc[2] = wmma_bf16(a, v2f, oacc[2]);
            oacc[3] = wmma_bf16(a, v3f, oacc[3]);
        }
    }

    // normalize and store to attn workspace [B,S,E] (bf16)
    #pragma unroll
    for (int r = 0; r < 8; ++r) {
        float inv = 1.0f / lrow[r];
        int s = q0 + wave * 16 + r + 8 * hf;
        #pragma unroll
        for (int nt = 0; nt < 4; ++nt) {
            int d = nt * 16 + nn;
            attn_ws[((size_t)(b * SEQ + s)) * EMBED + h * HDIM + d] = (__bf16)(oacc[nt][r] * inv);
        }
    }
}

// ---- Kernel 3: output projection ------------------------------------------
// out[8192,1024] = attn[8192,1024](bf16) @ w_proj[1024,1024] + b_proj  (fp32 out)
__global__ __launch_bounds__(256)
void out_proj_kernel(const __bf16* __restrict__ a_ws, const float* __restrict__ w,
                     const float* __restrict__ bias, float* __restrict__ out)
{
    __shared__ __bf16 As[128][72];
    __shared__ __bf16 Bts[128][72];
    const int tid = threadIdx.x, lane = tid & 31, wave = tid >> 5;
    const int wm = wave & 3, wn = wave >> 2;
    const int m0 = blockIdx.y * 128, n0 = blockIdx.x * 128;
    f32x8 acc[2][4] = {};

    for (int k0 = 0; k0 < EMBED; k0 += 64) {
        // stage A: bf16 pure copy -> async to LDS (4 x 16B per thread)
        #pragma unroll
        for (int i = 0; i < 4; ++i) {
            int idx8 = tid + i * 256;                    // 1024 chunks of 8 bf16
            int row = idx8 >> 3, c8 = (idx8 & 7) << 3;
            async_copy_b128(&As[row][c8], a_ws + (size_t)(m0 + row) * EMBED + k0 + c8);
        }
        // stage B transposed: 64x128 fp32 -> Bts[n][k]
        #pragma unroll
        for (int i = 0; i < 8; ++i) {
            int idx4 = tid + i * 256;
            int kr = idx4 >> 5, c4 = (idx4 & 31) << 2;
            f32x4 v = *(const f32x4*)(w + (size_t)(k0 + kr) * EMBED + n0 + c4);
            #pragma unroll
            for (int e = 0; e < 4; ++e) Bts[c4 + e][kr] = (__bf16)v[e];
        }
        if (k0 + 64 < EMBED)
            __builtin_prefetch(w + (size_t)(k0 + 64 + (tid >> 2)) * EMBED + n0 + (tid & 3) * 32, 0, 3);
        wait_async();
        __syncthreads();
        #pragma unroll
        for (int kk = 0; kk < 64; kk += 32) {
            bf16x16 a0 = load_frag_a(&As[wm * 32][kk], 72, lane);
            bf16x16 a1 = load_frag_a(&As[wm * 32 + 16][kk], 72, lane);
            bf16x16 b0 = load_frag_b(&Bts[wn * 64 + 0][kk], 72, lane);
            bf16x16 b1 = load_frag_b(&Bts[wn * 64 + 16][kk], 72, lane);
            bf16x16 b2 = load_frag_b(&Bts[wn * 64 + 32][kk], 72, lane);
            bf16x16 b3 = load_frag_b(&Bts[wn * 64 + 48][kk], 72, lane);
            acc[0][0] = wmma_bf16(a0, b0, acc[0][0]);
            acc[1][0] = wmma_bf16(a1, b0, acc[1][0]);
            acc[0][1] = wmma_bf16(a0, b1, acc[0][1]);
            acc[1][1] = wmma_bf16(a1, b1, acc[1][1]);
            acc[0][2] = wmma_bf16(a0, b2, acc[0][2]);
            acc[1][2] = wmma_bf16(a1, b2, acc[1][2]);
            acc[0][3] = wmma_bf16(a0, b3, acc[0][3]);
            acc[1][3] = wmma_bf16(a1, b3, acc[1][3]);
        }
        __syncthreads();
    }

    const int hf = lane >> 4, nn = lane & 15;
    #pragma unroll
    for (int mt = 0; mt < 2; ++mt) {
        #pragma unroll
        for (int nt = 0; nt < 4; ++nt) {
            int n = n0 + wn * 64 + nt * 16 + nn;
            float bb = bias[n];
            #pragma unroll
            for (int r = 0; r < 8; ++r) {
                int m = m0 + wm * 32 + mt * 16 + r + 8 * hf;
                out[(size_t)m * EMBED + n] = acc[mt][nt][r] + bb;
            }
        }
    }
}

// ---- launch ----------------------------------------------------------------
extern "C" void kernel_launch(void* const* d_in, const int* in_sizes, int n_in,
                              void* d_out, int out_size, void* d_ws, size_t ws_size,
                              hipStream_t stream) {
    const float* x      = (const float*)d_in[0];
    const float* w_qkv  = (const float*)d_in[1];
    const float* b_qkv  = (const float*)d_in[2];
    const float* w_proj = (const float*)d_in[3];
    const float* b_proj = (const float*)d_in[4];
    float* out = (float*)d_out;

    const size_t seg_bytes = (size_t)BATCH * NHEADS * SEQ * HDIM * sizeof(__bf16); // 16 MB
    __bf16* qkv_ws  = (__bf16*)d_ws;                                   // Q|K|V: 48 MB
    __bf16* attn_ws = (__bf16*)((char*)d_ws + 3 * seg_bytes);          // 16 MB

    dim3 g1(3072 / 128, (BATCH * SEQ) / 128);     // 24 x 64
    qkv_proj_kernel<<<g1, 256, 0, stream>>>(x, w_qkv, b_qkv, qkv_ws);

    dim3 g2(SEQ / 128, NHEADS, BATCH);            // 16 x 16 x 4
    attn_kernel<<<g2, 256, 0, stream>>>(qkv_ws, attn_ws);

    dim3 g3(EMBED / 128, (BATCH * SEQ) / 128);    // 8 x 64
    out_proj_kernel<<<g3, 256, 0, stream>>>(attn_ws, w_proj, b_proj, out);
}
